// NMTDecoder_17308718203638
// MI455X (gfx1250) — matrix operand order repaired and checked by reading, not compiled, not run on hardware
//
#include <hip/hip_runtime.h>
#include <hip/hip_bf16.h>

// ---------------------------------------------------------------------------
// NMT greedy decoder for gfx1250 (MI455X).  bf16 WMMA (16x16x32) GEMMs,
// everything converted to bf16 once so the full working set (~140MB) lives in
// the 192MB L2.  512 sequential steps, 4 kernels per step, all on `stream`.
// ---------------------------------------------------------------------------

#define BB 64
#define TT 512
#define HH 512
#define EE 128
#define VV 2048
#define XK 1664   // E + 3H  (LSTM input width)
#define YK 1024   // 2H      (output-proj input width)

typedef __attribute__((ext_vector_type(16))) __bf16 v16bf;
typedef __attribute__((ext_vector_type(8)))  __bf16 v8bf;
typedef __attribute__((ext_vector_type(4)))  __bf16 v4bf;
typedef __attribute__((ext_vector_type(2)))  __bf16 v2bf;
typedef __attribute__((ext_vector_type(8)))  float  v8f;
typedef __attribute__((ext_vector_type(4)))  float  v4f;

// ---- WMMA tile loaders (contiguous k = 16*(lane/16) + s in A and B, wave32) ------------------

// A-matrix 16x32 bf16: lane&15 = row M; lanes 0-15 hold K 0..15,
// lanes 16-31 hold K 16..31.  base points at (row0,k0), row-major.
__device__ __forceinline__ v16bf load_a_tile(const __bf16* base, int lda) {
  const int lane = threadIdx.x & 31;
  const __bf16* p = base + (lane & 15) * lda + ((lane >> 4) << 4);
  v8bf lo = *(const v8bf*)(p);
  v8bf hi = *(const v8bf*)(p + 8);
  return __builtin_shufflevector(lo, hi, 0, 1, 2, 3, 4, 5, 6, 7,
                                         8, 9, 10, 11, 12, 13, 14, 15);
}

// B-matrix 32x16 bf16 where B[k][n] = W[n][k] (we compute X @ W^T):
// lane&15 = column N (a row of W); lanes 0-15 hold K 0..15, lanes 16-31 K 16..31.
__device__ __forceinline__ v16bf load_bt_tile(const __bf16* base, int ldb) {
  const int lane = threadIdx.x & 31;
  const __bf16* p = base + (lane & 15) * ldb + ((lane >> 4) << 4);
  return *(const v16bf*)p;
}

__device__ __forceinline__ float sigmoidf_(float v) {
  return 1.0f / (1.0f + __expf(-v));
}

// ---- one-time prep ---------------------------------------------------------

// vectorized f32 -> bf16 (n must be a multiple of 4; all our tensors are)
__global__ void k_cvt(const float* __restrict__ src, __bf16* __restrict__ dst, int n4) {
  int i = blockIdx.x * blockDim.x + threadIdx.x;
  int stride = gridDim.x * blockDim.x;
  for (; i < n4; i += stride) {
    v4f s = ((const v4f*)src)[i];
    v4bf d;
#pragma unroll
    for (int j = 0; j < 4; ++j) d[j] = (__bf16)s[j];
    ((v4bf*)dst)[i] = d;
  }
}

__global__ void k_bias(const float* __restrict__ b_ih, const float* __restrict__ b_hh,
                       float* __restrict__ bias) {
  int i = blockIdx.x * blockDim.x + threadIdx.x;
  if (i < 4 * HH) bias[i] = b_ih[i] + b_hh[i];
}

// ctx0 = enc[b, len-1]; emb0 = 0; enc_0 staged into x.
__global__ __launch_bounds__(256) void k_init(const float* __restrict__ enc,
                                              const int* __restrict__ mask,
                                              __bf16* __restrict__ x,
                                              __bf16* __restrict__ y) {
  const int b = blockIdx.x;
  __shared__ int red[256];
  int s = 0;
  for (int t = threadIdx.x; t < TT; t += 256) s += mask[b * TT + t];
  red[threadIdx.x] = s;
  __syncthreads();
  for (int off = 128; off > 0; off >>= 1) {
    if (threadIdx.x < off) red[threadIdx.x] += red[threadIdx.x + off];
    __syncthreads();
  }
  int len = red[0];
  if (len < 1) len = 1;
  const float* c0 = enc + ((size_t)b * TT + (len - 1)) * HH;
  const float* e0 = enc + ((size_t)b * TT + 0) * HH;
  for (int h = threadIdx.x; h < HH; h += 256) {
    __bf16 cv = (__bf16)c0[h];
    x[b * XK + EE + h]      = cv;              // context slot of x
    y[b * YK + HH + h]      = cv;              // context slot of y
    x[b * XK + EE + HH + h] = (__bf16)e0[h];   // enc_t slot (t=0)
  }
  for (int e = threadIdx.x; e < EE; e += 256) x[b * XK + e] = (__bf16)0.0f;
}

// energies[b,t,:] = enc[b,t,:] @ W_attn^T + b_attn  (stored bf16)
// grid = (H/16=32, M/64=512), block = 128 (4 waves, one 16-row M-tile each)
__global__ __launch_bounds__(128) void k_energy(const __bf16* __restrict__ encb,
                                                const __bf16* __restrict__ Wattb,
                                                const float* __restrict__ battn,
                                                __bf16* __restrict__ energ) {
  const int n0 = blockIdx.x * 16;
  const int m0 = blockIdx.y * 64 + (threadIdx.x >> 5) * 16;
  const __bf16* A = encb + (size_t)m0 * HH;
  const __bf16* W = Wattb + (size_t)n0 * HH;
  v8f acc = {};
  for (int k0 = 0; k0 < HH; k0 += 32) {
    v16bf a = load_a_tile(A + k0, HH);
    v16bf b = load_bt_tile(W + k0, HH);
    acc = __builtin_amdgcn_wmma_f32_16x16x32_bf16(false, a, false, b,
                                                  (short)0, acc, false, false);
  }
  const int lane = threadIdx.x & 31;
  const int n = lane & 15;
  const int mrow = (lane >> 4) * 8;
  const float bv = battn[n0 + n];
#pragma unroll
  for (int r = 0; r < 8; ++r) {
    int m = m0 + mrow + r;
    energ[(size_t)m * HH + n0 + n] = (__bf16)(acc[r] + bv);
  }
}

// ---- per-step kernels ------------------------------------------------------

// gates = x @ W_ih^T + bias, but only i/g/o gates (f*c0 == 0), fused LSTM
// epilogue -> hdn (f32 for attention) + y[:, :H] (bf16 for score GEMM).
// grid = H/16 = 32, block = 128 (4 waves = 4 batch M-tiles).
__global__ __launch_bounds__(128) void k_gates(const __bf16* __restrict__ x,
                                               const __bf16* __restrict__ Wihb,
                                               const float* __restrict__ bias,
                                               float* __restrict__ hdn,
                                               __bf16* __restrict__ y) {
  const int n0 = blockIdx.x * 16;           // h columns
  const int m0 = (threadIdx.x >> 5) * 16;   // batch rows
  const __bf16* Wi = Wihb + (size_t)(0 * HH + n0) * XK;  // i-gate rows
  const __bf16* Wg = Wihb + (size_t)(2 * HH + n0) * XK;  // g-gate rows
  const __bf16* Wo = Wihb + (size_t)(3 * HH + n0) * XK;  // o-gate rows
  const __bf16* xa = x + (size_t)m0 * XK;
  v8f ai = {}, ag = {}, ao = {};
  for (int k0 = 0; k0 < XK; k0 += 32) {
    v16bf a  = load_a_tile(xa + k0, XK);     // shared A tile for 3 WMMAs
    v16bf bi = load_bt_tile(Wi + k0, XK);
    v16bf bg = load_bt_tile(Wg + k0, XK);
    v16bf bo = load_bt_tile(Wo + k0, XK);
    ai = __builtin_amdgcn_wmma_f32_16x16x32_bf16(false, a, false, bi, (short)0, ai, false, false);
    ag = __builtin_amdgcn_wmma_f32_16x16x32_bf16(false, a, false, bg, (short)0, ag, false, false);
    ao = __builtin_amdgcn_wmma_f32_16x16x32_bf16(false, a, false, bo, (short)0, ao, false, false);
  }
  const int lane = threadIdx.x & 31;
  const int n = lane & 15;
  const int mrow = (lane >> 4) * 8;
  const float bi_ = bias[0 * HH + n0 + n];
  const float bg_ = bias[2 * HH + n0 + n];
  const float bo_ = bias[3 * HH + n0 + n];
#pragma unroll
  for (int r = 0; r < 8; ++r) {
    const int b = m0 + mrow + r;
    float c = sigmoidf_(ai[r] + bi_) * tanhf(ag[r] + bg_);
    float h = sigmoidf_(ao[r] + bo_) * tanhf(c);
    hdn[b * HH + n0 + n] = h;
    y[b * YK + n0 + n] = (__bf16)h;
  }
}

// score = [hdn, ctx] @ W_out^T + b_out.  grid = V/16 = 128, block = 128.
__global__ __launch_bounds__(128) void k_score(const __bf16* __restrict__ y,
                                               const __bf16* __restrict__ Woutb,
                                               const float* __restrict__ bout,
                                               float* __restrict__ score) {
  const int n0 = blockIdx.x * 16;
  const int m0 = (threadIdx.x >> 5) * 16;
  const __bf16* W = Woutb + (size_t)n0 * YK;
  const __bf16* ya = y + (size_t)m0 * YK;
  v8f acc = {};
  for (int k0 = 0; k0 < YK; k0 += 32) {
    v16bf a = load_a_tile(ya + k0, YK);
    v16bf b = load_bt_tile(W + k0, YK);
    acc = __builtin_amdgcn_wmma_f32_16x16x32_bf16(false, a, false, b,
                                                  (short)0, acc, false, false);
  }
  const int lane = threadIdx.x & 31;
  const int n = lane & 15;
  const int mrow = (lane >> 4) * 8;
  const float bv = bout[n0 + n];
#pragma unroll
  for (int r = 0; r < 8; ++r) {
    const int b = m0 + mrow + r;
    score[b * VV + n0 + n] = acc[r] + bv;
  }
}

// log-softmax + argmax over V, write logp to output, gather next embedding.
__global__ __launch_bounds__(256) void k_softmax(const float* __restrict__ score,
                                                 const __bf16* __restrict__ embb,
                                                 float* __restrict__ out,
                                                 __bf16* __restrict__ x, int t) {
  const int b = blockIdx.x;
  const float* s = score + (size_t)b * VV;
  __shared__ float rmax[256];
  __shared__ float rsum[256];
  __shared__ int   ridx[256];
  float mx = -3.4e38f;
  int mi = 0;
  for (int v = threadIdx.x; v < VV; v += 256) {
    float sv = s[v];
    if (sv > mx) { mx = sv; mi = v; }
  }
  rmax[threadIdx.x] = mx;
  ridx[threadIdx.x] = mi;
  __syncthreads();
  for (int off = 128; off > 0; off >>= 1) {
    if (threadIdx.x < off) {
      if (rmax[threadIdx.x + off] > rmax[threadIdx.x]) {
        rmax[threadIdx.x] = rmax[threadIdx.x + off];
        ridx[threadIdx.x] = ridx[threadIdx.x + off];
      }
    }
    __syncthreads();
  }
  const float gmax = rmax[0];
  const int pred = ridx[0];
  float sum = 0.0f;
  for (int v = threadIdx.x; v < VV; v += 256) sum += __expf(s[v] - gmax);
  rsum[threadIdx.x] = sum;
  __syncthreads();
  for (int off = 128; off > 0; off >>= 1) {
    if (threadIdx.x < off) rsum[threadIdx.x] += rsum[threadIdx.x + off];
    __syncthreads();
  }
  const float lse = gmax + __logf(rsum[0]);
  float* o = out + ((size_t)b * TT + t) * VV;
  for (int v = threadIdx.x; v < VV; v += 256) o[v] = s[v] - lse;
  // greedy feedback: embedded_{t+1} = emb_table[pred]
  for (int e = threadIdx.x; e < EE; e += 256)
    x[b * XK + e] = embb[(size_t)pred * EE + e];
}

// attention: logits = energies . hdn (masked), softmax over T, ctx = alpha.enc;
// also stages enc_{t+1} into x for the next step.  grid = B, block = 256.
// All global traffic is vectorized: 16B v8bf reads in phase 1 (full cacheline
// per wave), packed 4B reads in phase 3 (two h columns per thread).
__global__ __launch_bounds__(256) void k_attn(const __bf16* __restrict__ energ,
                                              const __bf16* __restrict__ encb,
                                              const float* __restrict__ hdn,
                                              const int* __restrict__ mask,
                                              __bf16* __restrict__ x,
                                              __bf16* __restrict__ y, int t) {
  const int b = blockIdx.x;
  __shared__ float sh[HH];    // hdn[b,:]
  __shared__ float sl[TT];    // logits -> alpha
  __shared__ float red[256];
  for (int k = threadIdx.x; k < HH; k += 256) sh[k] = hdn[b * HH + k];
  __syncthreads();
  // phase 1: logits[tau] = <energies[b,tau,:], hdn[b,:]>, vectorized 8-wide
  for (int tau = threadIdx.x; tau < TT; tau += 256) {
    const v8bf* ev = (const v8bf*)(energ + ((size_t)b * TT + tau) * HH);
    float acc = 0.0f;
#pragma unroll 4
    for (int kv = 0; kv < HH / 8; ++kv) {
      v8bf e8 = ev[kv];
      const float* hv = &sh[kv * 8];
#pragma unroll
      for (int j = 0; j < 8; ++j) acc += (float)e8[j] * hv[j];
    }
    sl[tau] = (mask[b * TT + tau] == 0) ? -1e12f : acc;
  }
  __syncthreads();
  // phase 2: softmax over T in LDS
  float mx = -3.4e38f;
  for (int tau = threadIdx.x; tau < TT; tau += 256) mx = fmaxf(mx, sl[tau]);
  red[threadIdx.x] = mx;
  __syncthreads();
  for (int off = 128; off > 0; off >>= 1) {
    if (threadIdx.x < off) red[threadIdx.x] = fmaxf(red[threadIdx.x], red[threadIdx.x + off]);
    __syncthreads();
  }
  const float gm = red[0];
  float sum = 0.0f;
  for (int tau = threadIdx.x; tau < TT; tau += 256) {   // each thread owns its taus
    float a = __expf(sl[tau] - gm);
    sl[tau] = a;
    sum += a;
  }
  red[threadIdx.x] = sum;
  __syncthreads();
  for (int off = 128; off > 0; off >>= 1) {
    if (threadIdx.x < off) red[threadIdx.x] += red[threadIdx.x + off];
    __syncthreads();
  }
  const float inv = 1.0f / red[0];
  for (int tau = threadIdx.x; tau < TT; tau += 256) sl[tau] *= inv;
  __syncthreads();
  // phase 3: ctx[h] = sum_tau alpha[tau] * enc[b,tau,h]; 2 columns per thread
  {
    const int h2 = threadIdx.x * 2;                 // 256 threads * 2 = 512 = H
    const __bf16* ebase = encb + (size_t)b * TT * HH + h2;
    float a0 = 0.0f, a1 = 0.0f;
#pragma unroll 4
    for (int tau = 0; tau < TT; ++tau) {
      v2bf ep = *(const v2bf*)(ebase + (size_t)tau * HH);   // packed 4B load
      float al = sl[tau];
      a0 += al * (float)ep[0];
      a1 += al * (float)ep[1];
    }
    v2bf cv;
    cv[0] = (__bf16)a0;
    cv[1] = (__bf16)a1;
    *(v2bf*)(x + b * XK + EE + h2) = cv;            // context slot of x
    *(v2bf*)(y + b * YK + HH + h2) = cv;            // context slot of y
  }
  // stage enc_{t+1} into x (dword copies)
  if (t + 1 < TT) {
    const uint* src = (const uint*)(encb + ((size_t)b * TT + (t + 1)) * HH);
    uint* dst = (uint*)(x + b * XK + EE + HH);
    dst[threadIdx.x] = src[threadIdx.x];            // 256 threads * 4B = 1KB = H*bf16
  }
}

// ---------------------------------------------------------------------------

extern "C" void kernel_launch(void* const* d_in, const int* in_sizes, int n_in,
                              void* d_out, int out_size, void* d_ws, size_t ws_size,
                              hipStream_t stream) {
  (void)in_sizes; (void)n_in; (void)out_size; (void)ws_size;
  const float* enc    = (const float*)d_in[0];
  const int*   mask   = (const int*)d_in[1];
  const float* emb    = (const float*)d_in[2];
  const float* W_ih   = (const float*)d_in[3];
  const float* b_ih   = (const float*)d_in[4];
  const float* b_hh   = (const float*)d_in[5];
  const float* W_attn = (const float*)d_in[6];
  const float* b_attn = (const float*)d_in[7];
  const float* W_out  = (const float*)d_in[8];
  const float* b_out  = (const float*)d_in[9];
  float* out = (float*)d_out;

  char* ws = (char*)d_ws;
  size_t off = 0;
  auto alloc = [&](size_t bytes) -> char* {
    char* p = ws + off;
    off += (bytes + 255) & ~(size_t)255;
    return p;
  };
  __bf16* encb  = (__bf16*)alloc((size_t)BB * TT * HH * 2);   // 32 MB
  __bf16* energ = (__bf16*)alloc((size_t)BB * TT * HH * 2);   // 32 MB
  __bf16* Wihb  = (__bf16*)alloc((size_t)4 * HH * XK * 2);    // 6.8 MB
  __bf16* Woutb = (__bf16*)alloc((size_t)VV * YK * 2);        // 4 MB
  __bf16* Wattb = (__bf16*)alloc((size_t)HH * HH * 2);
  __bf16* embb  = (__bf16*)alloc((size_t)VV * EE * 2);
  float*  bias  = (float*)alloc((size_t)4 * HH * 4);
  __bf16* x     = (__bf16*)alloc((size_t)BB * XK * 2);
  __bf16* y     = (__bf16*)alloc((size_t)BB * YK * 2);
  float*  hdn   = (float*)alloc((size_t)BB * HH * 4);
  float*  score = (float*)alloc((size_t)BB * VV * 4);

  // one-time: bf16 conversions (vectorized x4), bias fold, init, energies GEMM
  k_cvt<<<4096, 256, 0, stream>>>(enc,    encb,  (BB * TT * HH) / 4);
  k_cvt<<<2048, 256, 0, stream>>>(W_ih,   Wihb,  (4 * HH * XK) / 4);
  k_cvt<<<1024, 256, 0, stream>>>(W_out,  Woutb, (VV * YK) / 4);
  k_cvt<<<256,  256, 0, stream>>>(W_attn, Wattb, (HH * HH) / 4);
  k_cvt<<<256,  256, 0, stream>>>(emb,    embb,  (VV * EE) / 4);
  k_bias<<<(4 * HH + 255) / 256, 256, 0, stream>>>(b_ih, b_hh, bias);
  k_init<<<BB, 256, 0, stream>>>(enc, mask, x, y);
  k_energy<<<dim3(HH / 16, (BB * TT) / 64), 128, 0, stream>>>(encb, Wattb, b_attn, energ);

  // 512 sequential decode steps
  for (int t = 0; t < TT; ++t) {
    k_gates<<<HH / 16, 128, 0, stream>>>(x, Wihb, bias, hdn, y);
    k_score<<<VV / 16, 128, 0, stream>>>(y, Woutb, b_out, score);
    k_softmax<<<BB, 256, 0, stream>>>(score, embb, out, x, t);
    k_attn<<<BB, 256, 0, stream>>>(energ, encb, hdn, mask, x, y, t);
  }
}
